// MambaBlock_27728308863667
// MI455X (gfx1250) — matrix operand order repaired and checked by reading, not compiled, not run on hardware
//
#include <hip/hip_runtime.h>

// ---------------------------------------------------------------- constants
#define BSZ    4
#define LSEQ   2304            // 48*48
#define DM     384
#define DI     768
#define NROW   (BSZ * LSEQ)    // 9216
#define DSTATE 16
#define DTR    24
#define XLD    64              // padded row stride of x_dbl (dt|B|C|pad)

typedef __attribute__((ext_vector_type(16))) _Float16 v16h;
typedef __attribute__((ext_vector_type(8)))  _Float16 v8h;
typedef __attribute__((ext_vector_type(8)))  float    v8f;

// ---------------------------------------------------------------- helpers
__device__ __forceinline__ float silu_f(float v) {
    return v / (1.f + __expf(-v));
}
__device__ __forceinline__ float softplus_f(float v) {
    return fmaxf(v, 0.f) + log1pf(__expf(-fabsf(v)));
}

__device__ __forceinline__ v8f wmma_f16(v16h a, v16h b, v8f c) {
    // D = A(16x32) * B(32x16) + C, f32 accumulate
    return __builtin_amdgcn_wmma_f32_16x16x32_f16(
        false, a, false, b, (short)0, c, false, false);
}

__device__ __forceinline__ v16h cat8(v8h g0, v8h g1) {
    return __builtin_shufflevector(g0, g1, 0, 1, 2, 3, 4, 5, 6, 7,
                                   8, 9, 10, 11, 12, 13, 14, 15);
}

// A fragment (f16 source): a[m][k], m = row0 + lane%16.
// element i: k = k0 + 16*half + i
__device__ __forceinline__ v16h frag_a_h(const _Float16* __restrict__ base, int ld,
                                         int row0, int k0) {
    const int lane = threadIdx.x & 31;
    const int half = lane >> 4;
    const _Float16* p = base + (size_t)(row0 + (lane & 15)) * ld + k0 + half * 16;
    return cat8(*(const v8h*)p, *(const v8h*)(p + 8));
}

// A fragment from f32 source (only used by the tiny dt GEMM on x_dbl).
__device__ __forceinline__ v16h frag_a_f32(const float* __restrict__ base, int ld,
                                           int row0, int k0) {
    const int lane = threadIdx.x & 31;
    const int half = lane >> 4;
    const float* p = base + (size_t)(row0 + (lane & 15)) * ld + k0 + half * 16;
    v8f g0 = *(const v8f*)p;
    v8f g1 = *(const v8f*)(p + 8);
    v16h r;
#pragma unroll
    for (int i = 0; i < 8; ++i) {
        r[i]     = (_Float16)g0[i];
        r[i + 8] = (_Float16)g1[i];
    }
    return r;
}

// B fragment (f16 weight, N x K row-major): b[k][n] = W[n][k].
// n = n0 + lane%16, element i: k = k0 + 16*half + i                (same order as frag_a_h)
__device__ __forceinline__ v16h frag_b_h(const _Float16* __restrict__ w, int ld,
                                         int n0, int k0) {
    const int lane = threadIdx.x & 31;
    const int half = lane >> 4;
    const _Float16* p = w + (size_t)(n0 + (lane & 15)) * ld + k0 + half * 16;
    return cat8(*(const v8h*)p, *(const v8h*)(p + 8));
}

// B fragment with row guard (rows >= nmax are zero), for N=56 padded to 64.
__device__ __forceinline__ v16h frag_b_h_guard(const _Float16* __restrict__ w, int ld,
                                               int n0, int k0, int nmax) {
    const int lane = threadIdx.x & 31;
    const int n = n0 + (lane & 15);
    if (n >= nmax) { v16h z = {}; return z; }
    const int half = lane >> 4;
    const _Float16* p = w + (size_t)n * ld + k0 + half * 16;
    return cat8(*(const v8h*)p, *(const v8h*)(p + 8));
}

// B fragment from dt_proj_w f16 (DI x 24): K padded 24->32, rows k>=24 zero.
__device__ __forceinline__ v16h frag_b_h_dt(const _Float16* __restrict__ w, int n0) {
    const int lane = threadIdx.x & 31;
    const int half = lane >> 4;
    const _Float16* p = w + (size_t)(n0 + (lane & 15)) * DTR + half * 16;
    v8h g0 = *(const v8h*)p;        // k = 16*half .. 16*half+7  (always < 24)
    v8h g1 = {};
    if (half == 0) g1 = *(const v8h*)(p + 8);   // k = 8..15 ; half==1 -> k>=24 zero
    return cat8(g0, g1);
}

// ---------------------------------------------------------------- prep: f32 -> f16
__global__ void k_cvt_h(const float* __restrict__ src, _Float16* __restrict__ dst, int n) {
    int i = blockIdx.x * 256 + threadIdx.x;
    if (i < n) dst[i] = (_Float16)src[i];
}

// ---------------------------------------------------------------- K0: x (b,c,l) -> xs_h (b*L, c) f16
__global__ void k_prep_xs(const float* __restrict__ x, _Float16* __restrict__ xs) {
    size_t i = (size_t)blockIdx.x * blockDim.x + threadIdx.x;
    if (i >= (size_t)NROW * DM) return;
    int c = (int)(i % DM);
    size_t r = i / DM;
    int b = (int)(r / LSEQ);
    int l = (int)(r % LSEQ);
    xs[i] = (_Float16)x[((size_t)b * DM + c) * LSEQ + l];
}

// ---------------------------------------------------------------- K1: xz = xs @ in_proj_w^T
// grid (288, 3), block 256. Wave -> 32(M) x 64(N) tile. Emits f16 xin / silu(z).
__global__ void k_inproj(const _Float16* __restrict__ xs, const _Float16* __restrict__ w,
                         _Float16* __restrict__ xinh, _Float16* __restrict__ zh) {
    const int wave = threadIdx.x >> 5, lane = threadIdx.x & 31;
    const int m0 = blockIdx.x * 32;
    const int n0 = (blockIdx.y * 8 + wave) * 64;
    v8f acc[2][4] = {};
    for (int k0 = 0; k0 < DM; k0 += 32) {
        if (k0 + 32 < DM)
            __builtin_prefetch(xs + (size_t)(m0 + (lane & 15)) * DM + k0 + 32, 0, 1);
        v16h a0 = frag_a_h(xs, DM, m0, k0);
        v16h a1 = frag_a_h(xs, DM, m0 + 16, k0);
#pragma unroll
        for (int j = 0; j < 4; ++j) {
            v16h b = frag_b_h(w, DM, n0 + j * 16, k0);
            acc[0][j] = wmma_f16(a0, b, acc[0][j]);
            acc[1][j] = wmma_f16(a1, b, acc[1][j]);
        }
    }
    const int mloc = 8 * (lane >> 4);
    const int nc = lane & 15;
#pragma unroll
    for (int mh = 0; mh < 2; ++mh)
#pragma unroll
        for (int j = 0; j < 4; ++j) {
            int n = n0 + j * 16 + nc;
#pragma unroll
            for (int r2 = 0; r2 < 8; ++r2) {
                size_t m = (size_t)m0 + 16 * mh + mloc + r2;
                float v = acc[mh][j][r2];
                if (n < DI) xinh[m * DI + n] = (_Float16)v;
                else        zh[m * DI + (n - DI)] = (_Float16)silu_f(v);
            }
        }
}

// ---------------------------------------------------------------- K2: causal depthwise conv + SiLU
// writes f32 (scan input) and f16 (x_proj GEMM input)
__global__ void k_conv(const _Float16* __restrict__ xinh, const float* __restrict__ cw,
                       const float* __restrict__ cb, float* __restrict__ xc,
                       _Float16* __restrict__ xch) {
    size_t i = (size_t)blockIdx.x * blockDim.x + threadIdx.x;
    if (i >= (size_t)NROW * DI) return;
    int d = (int)(i % DI);
    size_t r = i / DI;
    int l = (int)(r % LSEQ);
    float acc = cb[d];
#pragma unroll
    for (int j = 0; j < 4; ++j) {
        int lj = l - 3 + j;
        if (lj >= 0) acc += cw[d * 4 + j] * (float)xinh[(r - 3 + j) * DI + d];
    }
    float v = silu_f(acc);
    xc[i]  = v;
    xch[i] = (_Float16)v;
}

// ---------------------------------------------------------------- K3: x_dbl = xc @ x_proj_w^T  (N=56)
// grid 36, block 256. Wave -> 32-row tile, all 4 N-tiles (guarded).
__global__ void k_xproj(const _Float16* __restrict__ xch, const _Float16* __restrict__ w,
                        float* __restrict__ xdbl) {
    const int wave = threadIdx.x >> 5, lane = threadIdx.x & 31;
    const int m0 = (blockIdx.x * 8 + wave) * 32;
    v8f acc[2][4] = {};
    for (int k0 = 0; k0 < DI; k0 += 32) {
        v16h a0 = frag_a_h(xch, DI, m0, k0);
        v16h a1 = frag_a_h(xch, DI, m0 + 16, k0);
#pragma unroll
        for (int j = 0; j < 4; ++j) {
            v16h b = frag_b_h_guard(w, DI, j * 16, k0, DTR + 2 * DSTATE);
            acc[0][j] = wmma_f16(a0, b, acc[0][j]);
            acc[1][j] = wmma_f16(a1, b, acc[1][j]);
        }
    }
    const int mloc = 8 * (lane >> 4);
    const int nc = lane & 15;
#pragma unroll
    for (int mh = 0; mh < 2; ++mh)
#pragma unroll
        for (int j = 0; j < 4; ++j) {
            int n = j * 16 + nc;
            if (n >= DTR + 2 * DSTATE) continue;
#pragma unroll
            for (int r2 = 0; r2 < 8; ++r2)
                xdbl[((size_t)m0 + 16 * mh + mloc + r2) * XLD + n] = acc[mh][j][r2];
        }
}

// ---------------------------------------------------------------- K4: dt = softplus(x_dbl[:,:24] @ dt_proj_w^T + b)
// K padded 24->32 (zero B rows); single k-step. grid 864, block 256.
__global__ void k_dtproj(const float* __restrict__ xdbl, const _Float16* __restrict__ w,
                         const float* __restrict__ bias, float* __restrict__ dt) {
    const int wave = threadIdx.x >> 5, lane = threadIdx.x & 31;
    const int job = blockIdx.x * 8 + wave;          // 6912 jobs
    const int m0 = (job / 12) * 16;
    const int n0 = (job % 12) * 64;
    v16h a = frag_a_f32(xdbl, XLD, m0, 0);          // k 24..31 garbage * zero B
    v8f acc[4] = {{}, {}, {}, {}};
#pragma unroll
    for (int j = 0; j < 4; ++j)
        acc[j] = wmma_f16(a, frag_b_h_dt(w, n0 + j * 16), acc[j]);
    const int mloc = 8 * (lane >> 4);
    const int nc = lane & 15;
#pragma unroll
    for (int j = 0; j < 4; ++j) {
        int n = n0 + j * 16 + nc;
        float bb = bias[n];
#pragma unroll
        for (int r2 = 0; r2 < 8; ++r2)
            dt[((size_t)m0 + mloc + r2) * DI + n] = softplus_f(acc[j][r2] + bb);
    }
}

// ---------------------------------------------------------------- K5: selective scan (fused +x*D, *silu(z))
// grid (4, 3), block 256. One thread per (b, channel); h[16] in VGPRs.
// B_t/C_t broadcast staged with CDNA5 async-to-LDS (ASYNCcnt).
__global__ void k_scan(const float* __restrict__ Alog, const float* __restrict__ Dp,
                       const float* __restrict__ dt, const float* __restrict__ xdbl,
                       const _Float16* __restrict__ zh, const float* __restrict__ xc,
                       _Float16* __restrict__ yh) {
    __shared__ float sBC[2 * DSTATE];
    const int b = blockIdx.x;
    const int d = blockIdx.y * 256 + threadIdx.x;
    float Aa[DSTATE], h[DSTATE];
#pragma unroll
    for (int s = 0; s < DSTATE; ++s) {
        Aa[s] = -__expf(Alog[d * DSTATE + s]);
        h[s] = 0.f;
    }
    const float Dd = Dp[d];
    const unsigned ldsoff = (unsigned)(size_t)(&sBC[threadIdx.x & 31]);
    for (int l = 0; l < LSEQ; ++l) {
        size_t r = (size_t)b * LSEQ + l;
        __syncthreads();                          // prior-step readers done
        if (threadIdx.x < 2 * DSTATE) {
            const float* gp = xdbl + r * XLD + DTR + threadIdx.x;  // cols 24..55 = B|C
            asm volatile("global_load_async_to_lds_b32 %0, %1, off"
                         :: "v"(ldsoff), "v"((unsigned long long)(size_t)gp)
                         : "memory");
        }
        asm volatile("s_wait_asynccnt 0x0" ::: "memory");
        __syncthreads();                          // LDS visible to all waves
        float xv  = xc[r * DI + d];
        float dtv = dt[r * DI + d];
        float y = 0.f;
#pragma unroll
        for (int s = 0; s < DSTATE; ++s) {
            float dA = __expf(dtv * Aa[s]);
            h[s] = dA * h[s] + (dtv * sBC[s]) * xv;
            y += h[s] * sBC[DSTATE + s];
        }
        y = (y + xv * Dd) * (float)zh[r * DI + d];
        yh[r * DI + d] = (_Float16)y;
    }
}

// ---------------------------------------------------------------- K6: out = y @ out_proj_w^T (N=384)
// grid 216, block 256. Wave -> 32(M) x 64(N) tile.
__global__ void k_outproj(const _Float16* __restrict__ yh, const _Float16* __restrict__ w,
                          float* __restrict__ outp) {
    const int wave = threadIdx.x >> 5, lane = threadIdx.x & 31;
    const int job = blockIdx.x * 8 + wave;          // 1728 jobs
    const int m0 = (job / 6) * 32;
    const int n0 = (job % 6) * 64;
    v8f acc[2][4] = {};
    for (int k0 = 0; k0 < DI; k0 += 32) {
        if (k0 + 32 < DI)
            __builtin_prefetch(yh + (size_t)(m0 + (lane & 15)) * DI + k0 + 32, 0, 1);
        v16h a0 = frag_a_h(yh, DI, m0, k0);
        v16h a1 = frag_a_h(yh, DI, m0 + 16, k0);
#pragma unroll
        for (int j = 0; j < 4; ++j) {
            v16h bf = frag_b_h(w, DI, n0 + j * 16, k0);
            acc[0][j] = wmma_f16(a0, bf, acc[0][j]);
            acc[1][j] = wmma_f16(a1, bf, acc[1][j]);
        }
    }
    const int mloc = 8 * (lane >> 4);
    const int nc = lane & 15;
#pragma unroll
    for (int mh = 0; mh < 2; ++mh)
#pragma unroll
        for (int j = 0; j < 4; ++j) {
            int n = n0 + j * 16 + nc;
#pragma unroll
            for (int r2 = 0; r2 < 8; ++r2)
                outp[((size_t)m0 + 16 * mh + mloc + r2) * DM + n] = acc[mh][j][r2];
        }
}

// ---------------------------------------------------------------- K7: LayerNorm + transpose to (b,c,h,w)
// grid 1152, block 256: one wave per row of 384.
__global__ void k_ln(const float* __restrict__ outp, const float* __restrict__ gamma,
                     const float* __restrict__ beta, float* __restrict__ out) {
    const int wave = threadIdx.x >> 5, lane = threadIdx.x & 31;
    const int r = blockIdx.x * 8 + wave;
    if (r >= NROW) return;
    float v[12], s = 0.f, s2 = 0.f;
#pragma unroll
    for (int j = 0; j < 12; ++j) {
        float t = outp[(size_t)r * DM + lane + j * 32];
        v[j] = t; s += t; s2 += t * t;
    }
#pragma unroll
    for (int off = 16; off > 0; off >>= 1) {
        s  += __shfl_xor(s, off);
        s2 += __shfl_xor(s2, off);
    }
    const float mu = s * (1.f / DM);
    const float var = s2 * (1.f / DM) - mu * mu;
    const float rstd = rsqrtf(var + 1e-5f);
    const int b = r / LSEQ, l = r % LSEQ;
#pragma unroll
    for (int j = 0; j < 12; ++j) {
        int c = lane + j * 32;
        out[((size_t)b * DM + c) * LSEQ + l] = (v[j] - mu) * rstd * gamma[c] + beta[c];
    }
}

// ---------------------------------------------------------------- launch
extern "C" void kernel_launch(void* const* d_in, const int* in_sizes, int n_in,
                              void* d_out, int out_size, void* d_ws, size_t ws_size,
                              hipStream_t stream) {
    (void)in_sizes; (void)n_in; (void)out_size; (void)ws_size;
    const float* x         = (const float*)d_in[0];
    const float* in_proj_w = (const float*)d_in[1];
    const float* conv_w    = (const float*)d_in[2];
    const float* conv_b    = (const float*)d_in[3];
    const float* x_proj_w  = (const float*)d_in[4];
    const float* dt_proj_w = (const float*)d_in[5];
    const float* dt_proj_b = (const float*)d_in[6];
    const float* A_log     = (const float*)d_in[7];
    const float* Dvec      = (const float*)d_in[8];
    const float* out_proj_w= (const float*)d_in[9];
    const float* ln_gamma  = (const float*)d_in[10];
    const float* ln_beta   = (const float*)d_in[11];

    char* wp = (char*)d_ws;
    auto carve = [&](size_t bytes) {
        char* q = wp;
        wp += (bytes + 255) & ~(size_t)255;
        return (void*)q;
    };
    float*    bufA = (float*)   carve((size_t)NROW * DI * 4);   // dt, then out_pre
    float*    bufB = (float*)   carve((size_t)NROW * DI * 4);   // xc (f32, scan input)
    float*    xdbl = (float*)   carve((size_t)NROW * XLD * 4);  // dt_raw|B|C
    _Float16* xs_h = (_Float16*)carve((size_t)NROW * DM * 2);
    _Float16* xch  = (_Float16*)carve((size_t)NROW * DI * 2);
    _Float16* yh   = (_Float16*)carve((size_t)NROW * DI * 2);   // xin_raw f16, then y f16
    _Float16* zh   = (_Float16*)carve((size_t)NROW * DI * 2);   // silu(z) f16
    _Float16* wih  = (_Float16*)carve((size_t)2 * DI * DM * 2);
    _Float16* wxh  = (_Float16*)carve((size_t)(DTR + 2 * DSTATE) * DI * 2);
    _Float16* wdh  = (_Float16*)carve((size_t)DI * DTR * 2);
    _Float16* woh  = (_Float16*)carve((size_t)DM * DI * 2);

    // weight conversions (tiny)
    k_cvt_h<<<(2 * DI * DM + 255) / 256, 256, 0, stream>>>(in_proj_w, wih, 2 * DI * DM);
    k_cvt_h<<<((DTR + 2 * DSTATE) * DI + 255) / 256, 256, 0, stream>>>(x_proj_w, wxh,
                                                                       (DTR + 2 * DSTATE) * DI);
    k_cvt_h<<<(DI * DTR + 255) / 256, 256, 0, stream>>>(dt_proj_w, wdh, DI * DTR);
    k_cvt_h<<<(DM * DI + 255) / 256, 256, 0, stream>>>(out_proj_w, woh, DM * DI);

    k_prep_xs<<<((size_t)NROW * DM + 255) / 256, 256, 0, stream>>>(x, xs_h);
    k_inproj<<<dim3(NROW / 32, 3), 256, 0, stream>>>(xs_h, wih, yh /*xin f16*/, zh);
    k_conv<<<((size_t)NROW * DI + 255) / 256, 256, 0, stream>>>(yh, conv_w, conv_b, bufB, xch);
    k_xproj<<<NROW / 32 / 8, 256, 0, stream>>>(xch, wxh, xdbl);
    k_dtproj<<<(NROW / 16) * 12 / 8, 256, 0, stream>>>(xdbl, wdh, dt_proj_b, bufA);
    k_scan<<<dim3(BSZ, DI / 256), 256, 0, stream>>>(A_log, Dvec, bufA, xdbl, zh, bufB, yh);
    k_outproj<<<(NROW / 32) * 6 / 8, 256, 0, stream>>>(yh, woh, bufA);
    k_ln<<<NROW / 8, 256, 0, stream>>>(bufA, ln_gamma, ln_beta, (float*)d_out);
}
